// MultiHeadAttn_68659347194495
// MI455X (gfx1250) — matrix operand order repaired and checked by reading, not compiled, not run on hardware
//
#include <hip/hip_runtime.h>

// ---------------------------------------------------------------------------
// MI455X (gfx1250) MHA block: QKV GEMM -> RoPE -> flash attention -> out-proj
// + residual + LayerNorm.  Matmuls via v_wmma_f32_16x16x32_bf16 (wave32).
// Attention stages K/V tiles through LDS with async-to-LDS DMA (ASYNCcnt).
// GEMM K-loops are 2x-unrolled ping-pong double buffers (no register rotation).
// ---------------------------------------------------------------------------

typedef __bf16 bf16_t;
typedef __attribute__((ext_vector_type(16))) __bf16 v16bf;
typedef __attribute__((ext_vector_type(8)))  __bf16 v8bf;
typedef __attribute__((ext_vector_type(8)))  float  v8f;
typedef __attribute__((ext_vector_type(4)))  int    v4i;

constexpr int Bc   = 16;
constexpr int Sc   = 512;
constexpr int HID  = 512;
constexpr int NH   = 14;
constexpr int Dh   = 64;
constexpr int NHD  = NH * Dh;     // 896
constexpr int Mtot = Bc * Sc;     // 8192
constexpr float LN_EPS     = 1e-5f;
constexpr float ALPHA_C    = 1.0f;
constexpr float INV_SQRT_D = 0.125f;   // 1/sqrt(64)

#define DEV __device__ __forceinline__

#if defined(__AMDGCN__) && __has_builtin(__builtin_amdgcn_global_load_async_to_lds_b128) && \
    __has_builtin(__builtin_amdgcn_s_wait_asynccnt)
#define USE_ASYNC_LDS 1
#else
#define USE_ASYNC_LDS 0
#endif

DEV v8f wmma_bf16(v16bf a, v16bf b, v8f c) {
  return __builtin_amdgcn_wmma_f32_16x16x32_bf16(false, a, false, b, (short)0, c, false, false);
}

// A-fragment (16x32, MxK) from row-major [rows][K] bf16.
// lane<16: row=lane, elems 0..15 = K 0..15
// lane>=16: row=lane-16, elems 0..15 = K 16..31
DEV v16bf load_a_frag(const bf16_t* base, int ld, int lane) {
  const int r  = lane & 15;
  const int kh = (lane & 16) ? 16 : 0;
  const bf16_t* p = base + (size_t)r * ld + kh;
  v8bf lo = *(const v8bf*)(p);
  v8bf hi = *(const v8bf*)(p + 8);
  return __builtin_shufflevector(lo, hi, 0,1,2,3,4,5,6,7,8,9,10,11,12,13,14,15);
}

// B-fragment (32x16, KxN) given B^T stored row-major [N][K] bf16.
// lane<16: col=lane, elems 0..15 = K 0..15 ; lane>=16: col=lane-16, K 16..31
DEV v16bf load_b_frag(const bf16_t* base, int ld, int lane) {
  const int n  = lane & 15;
  const int kh = (lane & 16) ? 16 : 0;
  const bf16_t* p = base + (size_t)n * ld + kh;
  v8bf lo = *(const v8bf*)(p);
  v8bf hi = *(const v8bf*)(p + 8);
  return __builtin_shufflevector(lo, hi, 0,1,2,3,4,5,6,7,8,9,10,11,12,13,14,15);
}

#if USE_ASYNC_LDS
// One 16-byte async DMA chunk: global -> LDS, tracked by ASYNCcnt.
// Builtin prototype (from hipcc diagnostic): (v4i AS1*, v4i AS3*, imm, imm).
DEV void async_copy16(const bf16_t* g, bf16_t* l) {
  __builtin_amdgcn_global_load_async_to_lds_b128(
      (__attribute__((address_space(1))) v4i*)g,
      (__attribute__((address_space(3))) v4i*)l, 0, 0);
}
#endif

// ---------------------------------------------------------------------------
// Kernel 1: fp32 -> bf16 pack
// ---------------------------------------------------------------------------
__global__ __launch_bounds__(256) void k_pack_bf16(const float* __restrict__ in,
                                                   bf16_t* __restrict__ out, int n) {
  int i = blockIdx.x * 256 + threadIdx.x;
  if (i < n) out[i] = (bf16_t)in[i];
}

// ---------------------------------------------------------------------------
// Kernel 2: out[m][n] = sum_k A[m][k] * W[n][k] + bias[n]   (bf16 in/out, f32 acc)
// Wave tile 16M x 64N.  K-loop unrolled 2x as ping-pong buffers A/B so loads
// overlap WMMAs without register rotation.  Requires Ktot % 64 == 0.
// grid = (Mtiles/8, N/64), block = 256.
// ---------------------------------------------------------------------------
__global__ __launch_bounds__(256) void k_gemm_bias(const bf16_t* __restrict__ A,
                                                   const bf16_t* __restrict__ W,
                                                   const float*  __restrict__ bias,
                                                   bf16_t* __restrict__ out,
                                                   int Ktot, int ldo) {
  const int lane = threadIdx.x & 31;
  const int wave = threadIdx.x >> 5;
  const int m0 = (blockIdx.x * 8 + wave) * 16;
  const int n0 = blockIdx.y * 64;
  const bf16_t* Arow = A + (size_t)m0 * Ktot;
  const bf16_t* W0 = W + (size_t)(n0 +  0) * Ktot;
  const bf16_t* W1 = W + (size_t)(n0 + 16) * Ktot;
  const bf16_t* W2 = W + (size_t)(n0 + 32) * Ktot;
  const bf16_t* W3 = W + (size_t)(n0 + 48) * Ktot;
  v8f acc[4] = {};

  v16bf aA = load_a_frag(Arow, Ktot, lane);
  v16bf bA0 = load_b_frag(W0, Ktot, lane);
  v16bf bA1 = load_b_frag(W1, Ktot, lane);
  v16bf bA2 = load_b_frag(W2, Ktot, lane);
  v16bf bA3 = load_b_frag(W3, Ktot, lane);

  int kk = 0;
  for (; kk + 64 < Ktot; kk += 64) {
    // stage B-buffer (kk+32) while WMMAs consume A-buffer (kk)
    v16bf aB  = load_a_frag(Arow + kk + 32, Ktot, lane);
    v16bf bB0 = load_b_frag(W0 + kk + 32, Ktot, lane);
    v16bf bB1 = load_b_frag(W1 + kk + 32, Ktot, lane);
    v16bf bB2 = load_b_frag(W2 + kk + 32, Ktot, lane);
    v16bf bB3 = load_b_frag(W3 + kk + 32, Ktot, lane);
    acc[0] = wmma_bf16(aA, bA0, acc[0]);
    acc[1] = wmma_bf16(aA, bA1, acc[1]);
    acc[2] = wmma_bf16(aA, bA2, acc[2]);
    acc[3] = wmma_bf16(aA, bA3, acc[3]);
    // stage A-buffer (kk+64) while WMMAs consume B-buffer (kk+32)
    aA  = load_a_frag(Arow + kk + 64, Ktot, lane);
    bA0 = load_b_frag(W0 + kk + 64, Ktot, lane);
    bA1 = load_b_frag(W1 + kk + 64, Ktot, lane);
    bA2 = load_b_frag(W2 + kk + 64, Ktot, lane);
    bA3 = load_b_frag(W3 + kk + 64, Ktot, lane);
    acc[0] = wmma_bf16(aB, bB0, acc[0]);
    acc[1] = wmma_bf16(aB, bB1, acc[1]);
    acc[2] = wmma_bf16(aB, bB2, acc[2]);
    acc[3] = wmma_bf16(aB, bB3, acc[3]);
  }
  {  // tail: last 64 (kk == Ktot - 64)
    v16bf aB  = load_a_frag(Arow + kk + 32, Ktot, lane);
    v16bf bB0 = load_b_frag(W0 + kk + 32, Ktot, lane);
    v16bf bB1 = load_b_frag(W1 + kk + 32, Ktot, lane);
    v16bf bB2 = load_b_frag(W2 + kk + 32, Ktot, lane);
    v16bf bB3 = load_b_frag(W3 + kk + 32, Ktot, lane);
    acc[0] = wmma_bf16(aA, bA0, acc[0]);
    acc[1] = wmma_bf16(aA, bA1, acc[1]);
    acc[2] = wmma_bf16(aA, bA2, acc[2]);
    acc[3] = wmma_bf16(aA, bA3, acc[3]);
    acc[0] = wmma_bf16(aB, bB0, acc[0]);
    acc[1] = wmma_bf16(aB, bB1, acc[1]);
    acc[2] = wmma_bf16(aB, bB2, acc[2]);
    acc[3] = wmma_bf16(aB, bB3, acc[3]);
  }

  const int mh = (lane & 16) ? 8 : 0;
#pragma unroll
  for (int j = 0; j < 4; ++j) {
    int n = n0 + 16 * j + (lane & 15);
    float bv = bias[n];
#pragma unroll
    for (int e = 0; e < 8; ++e)
      out[(size_t)(m0 + mh + e) * ldo + n] = (bf16_t)(acc[j][e] + bv);
  }
}

// ---------------------------------------------------------------------------
// Kernel 3: RoPE on q,k and layout transform.
//   qraw/kraw/vraw : [B,S,NH*D] bf16   ->  q_r,k_r : [B,NH,S,D] ; v_t : [B,NH,D,S]
// one thread per (b,s,h,dpair), dpair in [0,32)
// ---------------------------------------------------------------------------
__global__ __launch_bounds__(256) void k_rope_pack(const bf16_t* __restrict__ qraw,
                                                   const bf16_t* __restrict__ kraw,
                                                   const bf16_t* __restrict__ vraw,
                                                   const float* __restrict__ pq,
                                                   const float* __restrict__ pk,
                                                   bf16_t* __restrict__ q_r,
                                                   bf16_t* __restrict__ k_r,
                                                   bf16_t* __restrict__ v_t) {
  int idx = blockIdx.x * 256 + threadIdx.x;
  if (idx >= Bc * Sc * NH * (Dh / 2)) return;
  const int dp = idx & 31;
  int t = idx >> 5;
  const int h = t % NH; t /= NH;
  const int s = t % Sc;
  const int b = t / Sc;
  const size_t ibase = ((size_t)(b * Sc + s)) * NHD + h * Dh;
  const size_t pbase = (((size_t)b * Sc + s) * NH + h) * Dh;
  const size_t obase = (((size_t)b * NH + h) * Sc + s) * Dh;
  const size_t vbase = (((size_t)b * NH + h) * Dh) * Sc;

  float q0v = (float)qraw[ibase + 2 * dp], q1v = (float)qraw[ibase + 2 * dp + 1];
  float k0v = (float)kraw[ibase + 2 * dp], k1v = (float)kraw[ibase + 2 * dp + 1];
  float sq = pq[pbase + dp], cq = pq[pbase + 32 + dp];
  float sk = pk[pbase + dp], ck = pk[pbase + 32 + dp];

  q_r[obase + dp]      = (bf16_t)(q0v * cq - q1v * sq);
  q_r[obase + 32 + dp] = (bf16_t)(q1v * cq + q0v * sq);
  k_r[obase + dp]      = (bf16_t)(k0v * ck - k1v * sk);
  k_r[obase + 32 + dp] = (bf16_t)(k1v * ck + k0v * sk);
  v_t[vbase + (size_t)(2 * dp) * Sc + s]     = vraw[ibase + 2 * dp];
  v_t[vbase + (size_t)(2 * dp + 1) * Sc + s] = vraw[ibase + 2 * dp + 1];
}

// ---------------------------------------------------------------------------
// Kernel 4: flash attention, transposed-score formulation.
//   S^T = K*Q^T (lane = query), online softmax = 8 in-lane + one xor-16.
//   O^T += V^T * P^T with P^T fragments built from score regs (no LDS).
//   K/V tiles cooperatively staged in double-buffered LDS via async DMA.
// grid = (S/128, NH, B), block 256 (8 waves x 16 query rows).
// ---------------------------------------------------------------------------
__global__ __launch_bounds__(256) void k_flash_attn(const bf16_t* __restrict__ q_r,
                                                    const bf16_t* __restrict__ k_r,
                                                    const bf16_t* __restrict__ v_t,
                                                    bf16_t* __restrict__ postx) {
  __shared__ __align__(16) bf16_t Ks[2][32 * 64];   // [local k][d]
  __shared__ __align__(16) bf16_t Vs[2][64 * 32];   // [d][local k]

  const int tid  = threadIdx.x;
  const int lane = tid & 31;
  const int wave = tid >> 5;
  const int b = blockIdx.z, h = blockIdx.y;
  const int q0 = blockIdx.x * 128 + wave * 16;

  const bf16_t* qh = q_r + ((size_t)(b * NH + h)) * Sc * Dh;
  const bf16_t* kh = k_r + ((size_t)(b * NH + h)) * Sc * Dh;
  const bf16_t* vh = v_t + ((size_t)(b * NH + h)) * Dh * Sc;

  // staging addresses for this thread (one 16B chunk of K, one of V^T)
  const int koff = tid * 8;                 // 256*8 = 32*64 elements
  const int vd   = tid >> 2;                // 0..63
  const int vc   = (tid & 3) * 8;           // 0,8,16,24
  const bf16_t* gK0 = kh + koff;            // + k0*Dh per tile
  const bf16_t* gV0 = vh + (size_t)vd * Sc + vc;  // + k0 per tile

  const v16bf bq0 = load_b_frag(qh + (size_t)q0 * Dh,      Dh, lane);
  const v16bf bq1 = load_b_frag(qh + (size_t)q0 * Dh + 32, Dh, lane);

  v8f o[4] = {};
  float mrun = -1e30f, lrun = 0.f;
  const bool hi = (lane & 16) != 0;
  constexpr int NT = Sc / 32;               // 16 k-tiles of 32

#if USE_ASYNC_LDS
  // prologue: DMA tile 0
  async_copy16(gK0, &Ks[0][koff]);
  async_copy16(gV0, &Vs[0][vd * 32 + vc]);
  __builtin_amdgcn_s_wait_asynccnt(0);
  __syncthreads();
#else
  {
    v8bf kreg = *(const v8bf*)gK0;
    v8bf vreg = *(const v8bf*)gV0;
    *(v8bf*)&Ks[0][koff]         = kreg;
    *(v8bf*)&Vs[0][vd * 32 + vc] = vreg;
    __syncthreads();
  }
#endif

  for (int kp = 0; kp < NT; ++kp) {
    const int cur = kp & 1;
#if USE_ASYNC_LDS
    if (kp + 1 < NT) {     // overlap next tile's DMA with this tile's WMMAs
      const int kn = (kp + 1) * 32;
      async_copy16(gK0 + (size_t)kn * Dh, &Ks[cur ^ 1][koff]);
      async_copy16(gV0 + kn,              &Vs[cur ^ 1][vd * 32 + vc]);
    }
#else
    v8bf kreg, vreg;
    if (kp + 1 < NT) {
      const int kn = (kp + 1) * 32;
      kreg = *(const v8bf*)(gK0 + (size_t)kn * Dh);
      vreg = *(const v8bf*)(gV0 + kn);
    }
#endif

    const bf16_t* Kb = Ks[cur];
    const bf16_t* Vb = Vs[cur];
    v8f s0 = {}, s1 = {};
    s0 = wmma_bf16(load_a_frag(Kb,                Dh, lane), bq0, s0);
    s0 = wmma_bf16(load_a_frag(Kb + 32,           Dh, lane), bq1, s0);
    s1 = wmma_bf16(load_a_frag(Kb + 16 * Dh,      Dh, lane), bq0, s1);
    s1 = wmma_bf16(load_a_frag(Kb + 16 * Dh + 32, Dh, lane), bq1, s1);

    float p0[8], p1[8];
    float tmax = -1e30f;
#pragma unroll
    for (int e = 0; e < 8; ++e) {
      p0[e] = s0[e] * INV_SQRT_D;
      p1[e] = s1[e] * INV_SQRT_D;
      tmax = fmaxf(tmax, fmaxf(p0[e], p1[e]));
    }
    tmax = fmaxf(tmax, __shfl_xor(tmax, 16, 32));
    const float mnew = fmaxf(mrun, tmax);
    const float corr = __expf(mrun - mnew);

    float psum = 0.f;
#pragma unroll
    for (int e = 0; e < 8; ++e) {
      p0[e] = __expf(p0[e] - mnew);
      p1[e] = __expf(p1[e] - mnew);
      psum += p0[e] + p1[e];
    }
    psum += __shfl_xor(psum, 16, 32);
    lrun = lrun * corr + psum;
    mrun = mnew;

#pragma unroll
    for (int f = 0; f < 4; ++f)
#pragma unroll
      for (int e = 0; e < 8; ++e) o[f][e] *= corr;

    // P^T B-fragment: lane<16 covers K=0..15 (tile 2kp), lane>=16 K=16..31
    v16bf pB;
#pragma unroll
    for (int e = 0; e < 8; ++e) {
      float t0 = __shfl_xor(p0[e], 16, 32);
      float t1 = __shfl_xor(p1[e], 16, 32);
      pB[e]     = (bf16_t)(hi ? t1 : p0[e]);
      pB[e + 8] = (bf16_t)(hi ? p1[e] : t0);
    }

#pragma unroll
    for (int f = 0; f < 4; ++f)
      o[f] = wmma_bf16(load_a_frag(Vb + f * 16 * 32, 32, lane), pB, o[f]);

#if USE_ASYNC_LDS
    __builtin_amdgcn_s_wait_asynccnt(0);
    __syncthreads();
#else
    __syncthreads();
    if (kp + 1 < NT) {
      *(v8bf*)&Ks[cur ^ 1][koff]         = kreg;
      *(v8bf*)&Vs[cur ^ 1][vd * 32 + vc] = vreg;
    }
    __syncthreads();
#endif
  }

  // O^T fragment: lane = query, vgprs = 8 consecutive d -> one 16B store each.
  const float invl = 1.f / lrun;
  const int mh = hi ? 8 : 0;
  const int qrow = q0 + (lane & 15);
  bf16_t* orow = postx + (size_t)(b * Sc + qrow) * NHD + h * Dh;
#pragma unroll
  for (int f = 0; f < 4; ++f) {
    v8bf ov;
#pragma unroll
    for (int e = 0; e < 8; ++e) ov[e] = (bf16_t)(o[f][e] * invl);
    *(v8bf*)(orow + f * 16 + mh) = ov;
  }
}

// ---------------------------------------------------------------------------
// Kernel 5: out = LN(x*alpha + postx @ Wo^T + bo)
// Block owns 16 rows x full 512 cols (8 waves x 64N), LDS tile + xor-shuffle LN.
// K-loop: same 2x-unrolled ping-pong as k_gemm_bias (896 = 14 * 64).
// ---------------------------------------------------------------------------
__global__ __launch_bounds__(256) void k_outproj_ln(const bf16_t* __restrict__ postx,
                                                    const bf16_t* __restrict__ wo,
                                                    const float* __restrict__ bo,
                                                    const float* __restrict__ x,
                                                    const float* __restrict__ gamma,
                                                    const float* __restrict__ beta,
                                                    float* __restrict__ out) {
  __shared__ float yt[16][520];
  const int lane = threadIdx.x & 31;
  const int wave = threadIdx.x >> 5;
  const int m0 = blockIdx.x * 16;
  const int n0 = wave * 64;
  const bf16_t* Arow = postx + (size_t)m0 * NHD;
  const bf16_t* W0 = wo + (size_t)(n0 +  0) * NHD;
  const bf16_t* W1 = wo + (size_t)(n0 + 16) * NHD;
  const bf16_t* W2 = wo + (size_t)(n0 + 32) * NHD;
  const bf16_t* W3 = wo + (size_t)(n0 + 48) * NHD;

  v8f acc[4] = {};
  v16bf aA  = load_a_frag(Arow, NHD, lane);
  v16bf bA0 = load_b_frag(W0, NHD, lane);
  v16bf bA1 = load_b_frag(W1, NHD, lane);
  v16bf bA2 = load_b_frag(W2, NHD, lane);
  v16bf bA3 = load_b_frag(W3, NHD, lane);

  int kk = 0;
  for (; kk + 64 < NHD; kk += 64) {
    v16bf aB  = load_a_frag(Arow + kk + 32, NHD, lane);
    v16bf bB0 = load_b_frag(W0 + kk + 32, NHD, lane);
    v16bf bB1 = load_b_frag(W1 + kk + 32, NHD, lane);
    v16bf bB2 = load_b_frag(W2 + kk + 32, NHD, lane);
    v16bf bB3 = load_b_frag(W3 + kk + 32, NHD, lane);
    acc[0] = wmma_bf16(aA, bA0, acc[0]);
    acc[1] = wmma_bf16(aA, bA1, acc[1]);
    acc[2] = wmma_bf16(aA, bA2, acc[2]);
    acc[3] = wmma_bf16(aA, bA3, acc[3]);
    aA  = load_a_frag(Arow + kk + 64, NHD, lane);
    bA0 = load_b_frag(W0 + kk + 64, NHD, lane);
    bA1 = load_b_frag(W1 + kk + 64, NHD, lane);
    bA2 = load_b_frag(W2 + kk + 64, NHD, lane);
    bA3 = load_b_frag(W3 + kk + 64, NHD, lane);
    acc[0] = wmma_bf16(aB, bB0, acc[0]);
    acc[1] = wmma_bf16(aB, bB1, acc[1]);
    acc[2] = wmma_bf16(aB, bB2, acc[2]);
    acc[3] = wmma_bf16(aB, bB3, acc[3]);
  }
  {
    v16bf aB  = load_a_frag(Arow + kk + 32, NHD, lane);
    v16bf bB0 = load_b_frag(W0 + kk + 32, NHD, lane);
    v16bf bB1 = load_b_frag(W1 + kk + 32, NHD, lane);
    v16bf bB2 = load_b_frag(W2 + kk + 32, NHD, lane);
    v16bf bB3 = load_b_frag(W3 + kk + 32, NHD, lane);
    acc[0] = wmma_bf16(aA, bA0, acc[0]);
    acc[1] = wmma_bf16(aA, bA1, acc[1]);
    acc[2] = wmma_bf16(aA, bA2, acc[2]);
    acc[3] = wmma_bf16(aA, bA3, acc[3]);
    acc[0] = wmma_bf16(aB, bB0, acc[0]);
    acc[1] = wmma_bf16(aB, bB1, acc[1]);
    acc[2] = wmma_bf16(aB, bB2, acc[2]);
    acc[3] = wmma_bf16(aB, bB3, acc[3]);
  }

  const int mh = (lane & 16) ? 8 : 0;
#pragma unroll
  for (int j = 0; j < 4; ++j) {
    int n = n0 + 16 * j + (lane & 15);
    float bv = bo[n];
#pragma unroll
    for (int e = 0; e < 8; ++e) yt[mh + e][n] = acc[j][e] + bv;
  }
  __syncthreads();

  const int r = threadIdx.x >> 4;   // 16 threads per row
  const int c = threadIdx.x & 15;
  const float* xrow = x + (size_t)(m0 + r) * HID;
  float sum = 0.f, sq = 0.f;
#pragma unroll
  for (int i = 0; i < 32; ++i) {
    int col = c + 16 * i;
    float v = yt[r][col] + ALPHA_C * xrow[col];
    yt[r][col] = v;
    sum += v; sq += v * v;
  }
#pragma unroll
  for (int msk = 8; msk >= 1; msk >>= 1) {   // xor<=8 stays in the 16-lane team
    sum += __shfl_xor(sum, msk, 32);
    sq  += __shfl_xor(sq,  msk, 32);
  }
  const float mu   = sum * (1.f / HID);
  const float var  = sq * (1.f / HID) - mu * mu;
  const float rstd = rsqrtf(var + LN_EPS);
  float* orow = out + (size_t)(m0 + r) * HID;
#pragma unroll
  for (int i = 0; i < 32; ++i) {
    int col = c + 16 * i;
    orow[col] = (yt[r][col] - mu) * rstd * gamma[col] + beta[col];
  }
}

// ---------------------------------------------------------------------------
extern "C" void kernel_launch(void* const* d_in, const int* in_sizes, int n_in,
                              void* d_out, int out_size, void* d_ws, size_t ws_size,
                              hipStream_t stream) {
  const float* x   = (const float*)d_in[0];
  const float* piq = (const float*)d_in[1];
  const float* pik = (const float*)d_in[2];
  // d_in[3]: src_key_padding_mask — all-True in this problem, no-op.
  const float* Wq = (const float*)d_in[4];
  const float* bq = (const float*)d_in[5];
  const float* Wk = (const float*)d_in[6];
  const float* bk = (const float*)d_in[7];
  const float* Wv = (const float*)d_in[8];
  const float* bv = (const float*)d_in[9];
  const float* Wo = (const float*)d_in[10];
  const float* bo = (const float*)d_in[11];
  const float* ga = (const float*)d_in[12];
  const float* be = (const float*)d_in[13];
  float* out = (float*)d_out;

  char* ws = (char*)d_ws;
  auto alloc = [&](size_t bytes) {
    char* p = ws;
    ws += (bytes + 255) & ~(size_t)255;
    return p;
  };
  bf16_t* xb   = (bf16_t*)alloc((size_t)Mtot * HID * sizeof(bf16_t));
  bf16_t* wqb  = (bf16_t*)alloc((size_t)NHD * HID * sizeof(bf16_t));
  bf16_t* wkb  = (bf16_t*)alloc((size_t)NHD * HID * sizeof(bf16_t));
  bf16_t* wvb  = (bf16_t*)alloc((size_t)NHD * HID * sizeof(bf16_t));
  bf16_t* wob  = (bf16_t*)alloc((size_t)HID * NHD * sizeof(bf16_t));
  bf16_t* qraw = (bf16_t*)alloc((size_t)Mtot * NHD * sizeof(bf16_t));
  bf16_t* kraw = (bf16_t*)alloc((size_t)Mtot * NHD * sizeof(bf16_t));
  bf16_t* vraw = (bf16_t*)alloc((size_t)Mtot * NHD * sizeof(bf16_t));
  bf16_t* q_r  = (bf16_t*)alloc((size_t)Mtot * NHD * sizeof(bf16_t));
  bf16_t* k_r  = (bf16_t*)alloc((size_t)Mtot * NHD * sizeof(bf16_t));
  bf16_t* v_t  = (bf16_t*)alloc((size_t)Mtot * NHD * sizeof(bf16_t));
  bf16_t* px   = (bf16_t*)alloc((size_t)Mtot * NHD * sizeof(bf16_t));

  // 1) pack fp32 -> bf16
  {
    int n = Mtot * HID;
    k_pack_bf16<<<(n + 255) / 256, 256, 0, stream>>>(x, xb, n);
    n = NHD * HID;
    k_pack_bf16<<<(n + 255) / 256, 256, 0, stream>>>(Wq, wqb, n);
    k_pack_bf16<<<(n + 255) / 256, 256, 0, stream>>>(Wk, wkb, n);
    k_pack_bf16<<<(n + 255) / 256, 256, 0, stream>>>(Wv, wvb, n);
    k_pack_bf16<<<(n + 255) / 256, 256, 0, stream>>>(Wo, wob, n);
  }

  // 2) QKV projections: (8192x896) = (8192x512) @ (896x512)^T + bias
  {
    dim3 grid(Mtot / 16 / 8, NHD / 64);
    k_gemm_bias<<<grid, 256, 0, stream>>>(xb, wqb, bq, qraw, HID, NHD);
    k_gemm_bias<<<grid, 256, 0, stream>>>(xb, wkb, bk, kraw, HID, NHD);
    k_gemm_bias<<<grid, 256, 0, stream>>>(xb, wvb, bv, vraw, HID, NHD);
  }

  // 3) RoPE + layout transforms
  {
    int n = Bc * Sc * NH * (Dh / 2);
    k_rope_pack<<<(n + 255) / 256, 256, 0, stream>>>(qraw, kraw, vraw, piq, pik,
                                                     q_r, k_r, v_t);
  }

  // 4) flash attention
  {
    dim3 grid(Sc / 128, NH, Bc);
    k_flash_attn<<<grid, 256, 0, stream>>>(q_r, k_r, v_t, px);
  }

  // 5) out projection + residual + LayerNorm
  k_outproj_ln<<<Mtot / 16, 256, 0, stream>>>(px, wob, bo, x, ga, be, out);

  (void)in_sizes; (void)n_in; (void)out_size; (void)ws_size;
}
